// MyLSTM_69080253989186
// MI455X (gfx1250) — matrix operand order repaired and checked by reading, not compiled, not run on hardware
//
#include <hip/hip_runtime.h>

typedef unsigned short ushort_t;
typedef __attribute__((ext_vector_type(16))) __bf16        v16bf;
typedef __attribute__((ext_vector_type(8)))  float         v8f;
typedef __attribute__((ext_vector_type(4)))  unsigned int  u32x4;

#define B_   32
#define S_   512
#define IN_  1024
#define H_   1024
#define OUT_ 1024
#define K_Z  (IN_ + H_)   // 2048

union FragCvt { u32x4 q[2]; v16bf v; };

// A-operand (16-bit 16x32): lanes 0-15 hold K 0-7 & 16-23, lanes 16-31 hold K 8-15 & 24-31.
__device__ __forceinline__ v16bf load_a_frag(const ushort_t* __restrict__ row, int k0, int sel) {
  const ushort_t* p = row + k0 + sel * 8;
  FragCvt c;
  c.q[0] = *reinterpret_cast<const u32x4*>(p);
  c.q[1] = *reinterpret_cast<const u32x4*>(p + 16);
  return c.v;
}

// B-operand (16-bit 32x16): lanes 0-15 hold K 0-7 & 16-23, lanes 16-31 hold K 8-15 & 24-31 (per N column row).
__device__ __forceinline__ v16bf load_b_frag(const ushort_t* __restrict__ row, int k0, int sel) {
  const ushort_t* p = row + k0 + sel * 8;
  FragCvt c;
  c.q[0] = *reinterpret_cast<const u32x4*>(p);
  c.q[1] = *reinterpret_cast<const u32x4*>(p + 16);
  return c.v;
}

__device__ __forceinline__ ushort_t f32_to_bf16_rne(float f) {
  unsigned u = __float_as_uint(f);
  u += 0x7FFFu + ((u >> 16) & 1u);
  return (ushort_t)(u >> 16);
}

__device__ __forceinline__ float sigmoidf_(float x) { return 1.0f / (1.0f + __expf(-x)); }

// ---------------- prep kernels ----------------

__global__ void cvt_f32_bf16_kernel(const float* __restrict__ src, ushort_t* __restrict__ dst, int n) {
  int i = blockIdx.x * blockDim.x + threadIdx.x;
  if (i < n) dst[i] = f32_to_bf16_rne(src[i]);
}

__global__ void init_state_kernel(ushort_t* __restrict__ h0, float* __restrict__ c0,
                                  float* __restrict__ biaspack,
                                  const float* __restrict__ bf, const float* __restrict__ bi,
                                  const float* __restrict__ bc, const float* __restrict__ bo) {
  int i = blockIdx.x * blockDim.x + threadIdx.x;
  if (i < B_ * H_) { h0[i] = 0; c0[i] = 0.0f; }
  if (i < H_) {
    biaspack[0 * H_ + i] = bf[i];
    biaspack[1 * H_ + i] = bi[i];
    biaspack[2 * H_ + i] = bc[i];
    biaspack[3 * H_ + i] = bo[i];
  }
}

// ---------------- one LSTM timestep ----------------
// grid = (H/16, B/16) = (64, 2); block = 128 threads = 4 waves (one per gate).
// Each wave runs TWO independent K-split WMMA accumulation chains to hide the
// WMMA->WMMA RAW latency on the sequential (latency-bound) recurrence.
__global__ __launch_bounds__(128) void lstm_step_kernel(
    const ushort_t* __restrict__ xbf,      // [B][S][IN] bf16
    const ushort_t* __restrict__ Wpack,    // [4][H][IN+H] bf16, gate order f,i,c,o, (N,K) row-major
    const float*    __restrict__ biaspack, // [4][H]
    const ushort_t* __restrict__ h_in,     // [B][H] bf16
    const float*    __restrict__ c_in,     // [B][H] f32
    ushort_t*       __restrict__ h_out,
    float*          __restrict__ c_out,
    ushort_t*       __restrict__ hs,       // [B][S][H] bf16 (collected h for final GEMM)
    int t)
{
  const int lane  = threadIdx.x & 31;
  const int gate  = threadIdx.x >> 5;      // 0=f 1=i 2=c 3=o
  const int sel   = lane >> 4;
  const int r     = lane & 15;
  const int nbase = blockIdx.x * 16;       // hidden-column tile
  const int mbase = blockIdx.y * 16;       // batch-row tile
  const int mrow  = mbase + r;

  const ushort_t* arow_h = h_in + (size_t)mrow * H_;
  const ushort_t* arow_x = xbf + ((size_t)mrow * S_ + t) * IN_;
  const ushort_t* brow   = Wpack + ((size_t)gate * H_ + nbase + r) * (size_t)K_Z;

  v8f acc0 = {}, acc1 = {};
  // K = 0..1023 comes from h (z = [h, x] concatenation order)
#pragma unroll 4
  for (int k0 = 0; k0 < H_; k0 += 64) {
    v16bf a0 = load_a_frag(arow_h, k0, sel);
    v16bf b0 = load_b_frag(brow, k0, sel);
    v16bf a1 = load_a_frag(arow_h, k0 + 32, sel);
    v16bf b1 = load_b_frag(brow, k0 + 32, sel);
    acc0 = __builtin_amdgcn_wmma_f32_16x16x32_bf16(false, a0, false, b0, (short)0, acc0, false, false);
    acc1 = __builtin_amdgcn_wmma_f32_16x16x32_bf16(false, a1, false, b1, (short)0, acc1, false, false);
  }
  // K = 1024..2047 comes from x_t
#pragma unroll 4
  for (int k0 = 0; k0 < IN_; k0 += 64) {
    v16bf a0 = load_a_frag(arow_x, k0, sel);
    v16bf b0 = load_b_frag(brow, H_ + k0, sel);
    v16bf a1 = load_a_frag(arow_x, k0 + 32, sel);
    v16bf b1 = load_b_frag(brow, H_ + k0 + 32, sel);
    acc0 = __builtin_amdgcn_wmma_f32_16x16x32_bf16(false, a0, false, b0, (short)0, acc0, false, false);
    acc1 = __builtin_amdgcn_wmma_f32_16x16x32_bf16(false, a1, false, b1, (short)0, acc1, false, false);
  }
  v8f acc = acc0 + acc1;

  // C/D layout: VGPR v -> row (v + 8*sel), col (lane&15)
  __shared__ float gbuf[4][16][16];
#pragma unroll
  for (int v = 0; v < 8; ++v)
    gbuf[gate][v + 8 * sel][r] = acc[v];
  __syncthreads();

  // fused elementwise: 256 cells, 128 threads -> 2 each (coalesced: e = j*128 + tid)
#pragma unroll
  for (int j = 0; j < 2; ++j) {
    int e = j * 128 + threadIdx.x;  // 0..255
    int m = e >> 4;
    int n = e & 15;
    int col  = nbase + n;
    int grow = mbase + m;
    float pf = gbuf[0][m][n] + biaspack[0 * H_ + col];
    float pi = gbuf[1][m][n] + biaspack[1 * H_ + col];
    float pc = gbuf[2][m][n] + biaspack[2 * H_ + col];
    float po = gbuf[3][m][n] + biaspack[3 * H_ + col];
    float fg = sigmoidf_(pf);
    float ig = sigmoidf_(pi);
    float cg = tanhf(pc);
    float og = sigmoidf_(po);
    float cold = c_in[(size_t)grow * H_ + col];
    float cnew = ig * cg + fg * cold;
    float hnew = og * tanhf(cnew);
    c_out[(size_t)grow * H_ + col] = cnew;
    ushort_t hb = f32_to_bf16_rne(hnew);
    h_out[(size_t)grow * H_ + col] = hb;
    hs[((size_t)grow * S_ + t) * H_ + col] = hb;
  }
}

// ---------------- final projection GEMM ----------------
// out[M=16384][N=1024] = hs[M][K=1024] * Wfc(N,K)^T + bfc
// grid = (M/32, N/64); block = 128 (4 waves, one 16-col N slice each, 2 M-tiles per wave)
__global__ __launch_bounds__(128) void final_gemm_kernel(
    const ushort_t* __restrict__ hs,
    const ushort_t* __restrict__ Wfc,   // [OUT][H] bf16
    const float*    __restrict__ bfc,
    float* __restrict__ out)
{
  const int lane  = threadIdx.x & 31;
  const int wid   = threadIdx.x >> 5;
  const int sel   = lane >> 4;
  const int r     = lane & 15;
  const int mbase = blockIdx.x * 32;
  const int nbase = blockIdx.y * 64 + wid * 16;

  const ushort_t* arow0 = hs + (size_t)(mbase + r) * H_;
  const ushort_t* arow1 = hs + (size_t)(mbase + 16 + r) * H_;
  const ushort_t* brow  = Wfc + (size_t)(nbase + r) * H_;

  v8f acc0 = {}, acc1 = {};
#pragma unroll 4
  for (int k0 = 0; k0 < H_; k0 += 32) {
    v16bf a0 = load_a_frag(arow0, k0, sel);
    v16bf a1 = load_a_frag(arow1, k0, sel);
    v16bf b  = load_b_frag(brow, k0, sel);
    acc0 = __builtin_amdgcn_wmma_f32_16x16x32_bf16(false, a0, false, b, (short)0, acc0, false, false);
    acc1 = __builtin_amdgcn_wmma_f32_16x16x32_bf16(false, a1, false, b, (short)0, acc1, false, false);
  }

  const int n = nbase + r;
  const float bias = bfc[n];
#pragma unroll
  for (int v = 0; v < 8; ++v) {
    int m0 = mbase + v + 8 * sel;
    out[(size_t)m0 * OUT_ + n]        = acc0[v] + bias;
    out[(size_t)(m0 + 16) * OUT_ + n] = acc1[v] + bias;
  }
}

// ---------------- launcher ----------------

extern "C" void kernel_launch(void* const* d_in, const int* in_sizes, int n_in,
                              void* d_out, int out_size, void* d_ws, size_t ws_size,
                              hipStream_t stream) {
  (void)in_sizes; (void)n_in; (void)out_size; (void)ws_size;
  const float* x    = (const float*)d_in[0];
  const float* Wf   = (const float*)d_in[1];
  const float* bf   = (const float*)d_in[2];
  const float* Wi   = (const float*)d_in[3];
  const float* bi   = (const float*)d_in[4];
  const float* Wc   = (const float*)d_in[5];
  const float* bc   = (const float*)d_in[6];
  const float* Wo   = (const float*)d_in[7];
  const float* bo   = (const float*)d_in[8];
  const float* Wfc  = (const float*)d_in[9];
  const float* bfc  = (const float*)d_in[10];
  float* out = (float*)d_out;

  char* ws = (char*)d_ws;
  size_t off = 0;
  auto alloc = [&](size_t bytes) -> char* {
    char* p = ws + off;
    off = (off + bytes + 255) & ~(size_t)255;
    return p;
  };
  ushort_t* xbf    = (ushort_t*)alloc((size_t)B_ * S_ * IN_ * 2);
  ushort_t* Wpack  = (ushort_t*)alloc((size_t)4 * H_ * K_Z * 2);
  ushort_t* Wfcbf  = (ushort_t*)alloc((size_t)OUT_ * H_ * 2);
  float*    biasp  = (float*)alloc((size_t)4 * H_ * 4);
  ushort_t* hping0 = (ushort_t*)alloc((size_t)B_ * H_ * 2);
  ushort_t* hping1 = (ushort_t*)alloc((size_t)B_ * H_ * 2);
  float*    cping0 = (float*)alloc((size_t)B_ * H_ * 4);
  float*    cping1 = (float*)alloc((size_t)B_ * H_ * 4);
  ushort_t* hsbuf  = (ushort_t*)alloc((size_t)B_ * S_ * H_ * 2);

  const int CT = 256;
  const int nx = B_ * S_ * IN_;
  const int nw = H_ * K_Z;
  const int nfc = OUT_ * H_;
  cvt_f32_bf16_kernel<<<(nx + CT - 1) / CT, CT, 0, stream>>>(x, xbf, nx);
  cvt_f32_bf16_kernel<<<(nw + CT - 1) / CT, CT, 0, stream>>>(Wf, Wpack + (size_t)0 * nw, nw);
  cvt_f32_bf16_kernel<<<(nw + CT - 1) / CT, CT, 0, stream>>>(Wi, Wpack + (size_t)1 * nw, nw);
  cvt_f32_bf16_kernel<<<(nw + CT - 1) / CT, CT, 0, stream>>>(Wc, Wpack + (size_t)2 * nw, nw);
  cvt_f32_bf16_kernel<<<(nw + CT - 1) / CT, CT, 0, stream>>>(Wo, Wpack + (size_t)3 * nw, nw);
  cvt_f32_bf16_kernel<<<(nfc + CT - 1) / CT, CT, 0, stream>>>(Wfc, Wfcbf, nfc);
  init_state_kernel<<<(B_ * H_ + CT - 1) / CT, CT, 0, stream>>>(hping0, cping0, biasp, bf, bi, bc, bo);

  dim3 stepGrid(H_ / 16, B_ / 16);  // (64, 2)
  for (int t = 0; t < S_; ++t) {
    const ushort_t* hin = (t & 1) ? hping1 : hping0;
    ushort_t*      hout = (t & 1) ? hping0 : hping1;
    const float*    cin = (t & 1) ? cping1 : cping0;
    float*         cout = (t & 1) ? cping0 : cping1;
    lstm_step_kernel<<<stepGrid, 128, 0, stream>>>(xbf, Wpack, biasp, hin, cin, hout, cout, hsbuf, t);
  }

  final_gemm_kernel<<<dim3((B_ * S_) / 32, OUT_ / 64), 128, 0, stream>>>(hsbuf, Wfcbf, bfc, out);
}
